// RGCNDistMult_80874234184062
// MI455X (gfx1250) — matrix operand order repaired and checked by reading, not compiled, not run on hardware
//
#include <hip/hip_runtime.h>
#include <hip/hip_bf16.h>

#define N_ENT    100000
#define N_REL    24
#define N_BASES  8
#define HID      256
#define N_EDGES  1000000
#define N_TRIPLES 65536

#define K_AGG    (N_BASES * HID)       // 2048
#define K_TOT    (K_AGG + HID)         // 2304 (basis stack + root)

typedef __attribute__((ext_vector_type(16))) __bf16          v16bf;
typedef __attribute__((ext_vector_type(8)))  float           v8f;
typedef __attribute__((ext_vector_type(16))) unsigned short  v16u;
typedef __attribute__((ext_vector_type(8)))  unsigned short  v8u;

// ---------- helpers ----------
__device__ __forceinline__ unsigned short b16(float f) {
    return __builtin_bit_cast(unsigned short, (__bf16)f);   // RNE, hw cvt
}

__device__ __forceinline__ v16bf combine16(v8u lo, v8u hi) {
    v16u u;
#pragma unroll
    for (int j = 0; j < 8; ++j) { u[j] = lo[j]; u[8 + j] = hi[j]; }
    return __builtin_bit_cast(v16bf, u);
}

// ---------- kernels ----------

// Per-(dst, rel) in-degree counts.
__global__ __launch_bounds__(256)
void rgcn_count_kernel(const int* __restrict__ edge_index,
                       const int* __restrict__ edge_type,
                       int* __restrict__ cnt) {
    int e = blockIdx.x * blockDim.x + threadIdx.x;
    if (e >= N_EDGES) return;
    int dst = edge_index[N_EDGES + e];
    int et  = edge_type[e];
    atomicAdd(&cnt[dst * N_REL + et], 1);
}

// Build bf16 transposed weight panel: wt[j*K_TOT + k]
//   k <  2048 : basis[b][i][j]  (k = b*256+i)   -> basis[k*256 + j]
//   k >= 2048 : root[k-2048][j]
__global__ __launch_bounds__(256)
void rgcn_build_wt_kernel(const float* __restrict__ basis,
                          const float* __restrict__ root,
                          unsigned short* __restrict__ wt) {
    int idx = blockIdx.x * blockDim.x + threadIdx.x;
    if (idx >= HID * K_TOT) return;
    int j = idx / K_TOT;
    int k = idx % K_TOT;
    float v = (k < K_AGG) ? basis[(size_t)k * HID + j]
                          : root[(size_t)(k - K_AGG) * HID + j];
    wt[idx] = b16(v);
}

// One 256-thread block per edge; thread d handles feature dim d.
__global__ __launch_bounds__(256)
void rgcn_scatter_kernel(const float* __restrict__ x,
                         const int* __restrict__ edge_index,
                         const int* __restrict__ edge_type,
                         const int* __restrict__ cnt,
                         const float* __restrict__ comp,
                         float* __restrict__ agg) {
    int e = blockIdx.x;
    int d = threadIdx.x;
    int src = edge_index[e];
    int dst = edge_index[N_EDGES + e];
    int et  = edge_type[e];
    float norm = 1.0f / fmaxf((float)cnt[dst * N_REL + et], 1.0f);
    float xv = x[(size_t)src * HID + d] * norm;
    float* aggd = agg + (size_t)dst * K_AGG + d;
#pragma unroll
    for (int b = 0; b < N_BASES; ++b) {
        atomicAdd(aggd + b * HID, comp[et * N_BASES + b] * xv);
    }
}

// out[M=100000, 256] = relu(bias + [agg | x] @ wt^T), bf16 WMMA, f32 accum.
// Block: 256 thr = 8 waves (2 M-waves x 4 N-waves); block tile M=64, N=256 (full).
// A tile staged through LDS as bf16 (each element converted once per block).
__global__ __launch_bounds__(256)
void rgcn_gemm_kernel(const float* __restrict__ agg,
                      const float* __restrict__ xin,
                      const unsigned short* __restrict__ wt,
                      const float* __restrict__ bias,
                      float* __restrict__ out) {
    __shared__ unsigned short lA[64][40];   // 64 rows x 32 cols bf16, stride 40 (80B)

    const int tid    = threadIdx.x;
    const int wid    = tid >> 5;          // 0..7
    const int lane   = tid & 31;
    const int half   = lane >> 4;         // 0 or 1
    const int l15    = lane & 15;
    const int wave_m = wid >> 2;          // 0..1
    const int wave_n = wid & 3;           // 0..3
    const int half8  = half * 8;

    const int block_row = blockIdx.x * 64;
    const int rowbase   = block_row + wave_m * 32;

    // staging assignment: thread -> (row, 8-col segment)
    const int st_row = tid >> 2;          // 0..63
    const int st_seg = (tid & 3) * 8;     // 0,8,16,24
    int st_grow = block_row + st_row;
    if (st_grow >= N_ENT) st_grow = N_ENT - 1;   // clamp tail (stores guarded)

    v8f acc[2][4];
#pragma unroll
    for (int rt = 0; rt < 2; ++rt)
#pragma unroll
        for (int ct = 0; ct < 4; ++ct)
            acc[rt][ct] = (v8f)0.0f;

    for (int kk = 0; kk < K_TOT; kk += 32) {
        // ---- stage A tile (64x32 f32 -> bf16 in LDS), one cvt per element ----
        const float* rp;
        if (kk < K_AGG)
            rp = agg + (size_t)st_grow * K_AGG + kk + st_seg;
        else
            rp = xin + (size_t)st_grow * HID + (kk - K_AGG) + st_seg;
        float4 f0 = *(const float4*)(rp + 0);
        float4 f1 = *(const float4*)(rp + 4);
        v8u u;
        u[0] = b16(f0.x); u[1] = b16(f0.y); u[2] = b16(f0.z); u[3] = b16(f0.w);
        u[4] = b16(f1.x); u[5] = b16(f1.y); u[6] = b16(f1.z); u[7] = b16(f1.w);
        *(v8u*)&lA[st_row][st_seg] = u;
        __syncthreads();

        // ---- A fragments from LDS ----
        // ISA 16-bit A 16x32 layout: lanes 0-15 (row=l15): K {0..7,16..23};
        // lanes 16-31: K {8..15,24..31}.
        v16bf a[2];
#pragma unroll
        for (int rt = 0; rt < 2; ++rt) {
            int rl = wave_m * 32 + rt * 16 + l15;
            v8u lo = *(const v8u*)&lA[rl][half8];
            v8u hi = *(const v8u*)&lA[rl][16 + half8];
            a[rt] = combine16(lo, hi);
        }

        // ---- B fragments from bf16 weights (row = output col, contiguous K) ----
        // ISA 16-bit B 32x16 layout: lanes 0-15 (col=l15): K {0..7,16..23};
        // lanes 16-31: K {8..15,24..31}.
        v16bf b[4];
#pragma unroll
        for (int ct = 0; ct < 4; ++ct) {
            int col = wave_n * 64 + ct * 16 + l15;
            const v8u* p = (const v8u*)(wt + (size_t)col * K_TOT + kk + half8);
            b[ct] = combine16(p[0], p[2]);
        }

#pragma unroll
        for (int rt = 0; rt < 2; ++rt)
#pragma unroll
            for (int ct = 0; ct < 4; ++ct)
                acc[rt][ct] = __builtin_amdgcn_wmma_f32_16x16x32_bf16(
                    false, a[rt], false, b[ct], (short)0, acc[rt][ct],
                    false, false);
        __syncthreads();
    }

    // Epilogue: bias + relu + store. C layout: VGPR v -> row = base + v + 8*half.
#pragma unroll
    for (int rt = 0; rt < 2; ++rt) {
#pragma unroll
        for (int ct = 0; ct < 4; ++ct) {
            int col = wave_n * 64 + ct * 16 + l15;
            float bval = bias[col];
            int rbase = rowbase + rt * 16 + half8;
#pragma unroll
            for (int v = 0; v < 8; ++v) {
                int row = rbase + v;
                if (row < N_ENT) {
                    float r = acc[rt][ct][v] + bval;
                    out[(size_t)row * HID + col] = r > 0.0f ? r : 0.0f;
                }
            }
        }
    }
}

// DistMult: one wave (32 lanes) per triple; lane handles 8 dims.
__global__ __launch_bounds__(256)
void rgcn_score_kernel(const float* __restrict__ x,
                       const float* __restrict__ rel,
                       const int* __restrict__ heads,
                       const int* __restrict__ rels,
                       const int* __restrict__ tails,
                       float* __restrict__ out) {
    int t = (blockIdx.x * blockDim.x + threadIdx.x) >> 5;   // uniform per wave
    int lane = threadIdx.x & 31;
    if (t >= N_TRIPLES) return;
    const float4* h  = (const float4*)(x   + (size_t)heads[t] * HID);
    const float4* tl = (const float4*)(x   + (size_t)tails[t] * HID);
    const float4* r  = (const float4*)(rel + (size_t)rels[t]  * HID);
    float4 h0 = h[lane * 2],  h1 = h[lane * 2 + 1];
    float4 r0 = r[lane * 2],  r1 = r[lane * 2 + 1];
    float4 t0 = tl[lane * 2], t1 = tl[lane * 2 + 1];
    float s = h0.x * r0.x * t0.x + h0.y * r0.y * t0.y +
              h0.z * r0.z * t0.z + h0.w * r0.w * t0.w +
              h1.x * r1.x * t1.x + h1.y * r1.y * t1.y +
              h1.z * r1.z * t1.z + h1.w * r1.w * t1.w;
#pragma unroll
    for (int m = 16; m > 0; m >>= 1) s += __shfl_xor(s, m, 32);
    if (lane == 0) out[t] = s;
}

// ---------- workspace layout (bytes, 256-aligned) ----------
#define CNT_BYTES   ((size_t)N_ENT * N_REL * sizeof(int))          //   9,600,000
#define AGG_BYTES   ((size_t)N_ENT * K_AGG * sizeof(float))        // 819,200,000
#define X_BYTES     ((size_t)N_ENT * HID * sizeof(float))          // 102,400,000
#define WT_BYTES    ((size_t)HID * K_TOT * sizeof(unsigned short)) //   1,179,648

#define OFF_CNT  ((size_t)0)
#define OFF_AGG  (OFF_CNT + CNT_BYTES)
#define OFF_X1   (OFF_AGG + AGG_BYTES)
#define OFF_X2   (OFF_X1 + X_BYTES)
#define OFF_W1   (OFF_X2 + X_BYTES)
#define OFF_W2   (OFF_W1 + ((WT_BYTES + 255) & ~(size_t)255))

extern "C" void kernel_launch(void* const* d_in, const int* in_sizes, int n_in,
                              void* d_out, int out_size, void* d_ws, size_t ws_size,
                              hipStream_t stream) {
    const float* entity_emb = (const float*)d_in[0];
    const float* rel_emb    = (const float*)d_in[1];
    const float* basis1     = (const float*)d_in[2];
    const float* comp1      = (const float*)d_in[3];
    const float* root1      = (const float*)d_in[4];
    const float* bias1      = (const float*)d_in[5];
    const float* basis2     = (const float*)d_in[6];
    const float* comp2      = (const float*)d_in[7];
    const float* root2      = (const float*)d_in[8];
    const float* bias2      = (const float*)d_in[9];
    const int* edge_index   = (const int*)d_in[10];
    const int* edge_type    = (const int*)d_in[11];
    const int* heads        = (const int*)d_in[12];
    const int* rels         = (const int*)d_in[13];
    const int* tails        = (const int*)d_in[14];
    float* out = (float*)d_out;

    char* ws = (char*)d_ws;
    int*   cnt = (int*)(ws + OFF_CNT);
    float* agg = (float*)(ws + OFF_AGG);
    float* x1  = (float*)(ws + OFF_X1);
    float* x2  = (float*)(ws + OFF_X2);
    unsigned short* wt1 = (unsigned short*)(ws + OFF_W1);
    unsigned short* wt2 = (unsigned short*)(ws + OFF_W2);

    const int GEMM_BLOCKS = (N_ENT + 63) / 64;   // 1563
    const int WT_ELEMS = HID * K_TOT;

    // degree counts (shared by both layers)
    hipMemsetAsync(cnt, 0, CNT_BYTES, stream);
    rgcn_count_kernel<<<(N_EDGES + 255) / 256, 256, 0, stream>>>(edge_index, edge_type, cnt);

    // bf16 transposed weight panels
    rgcn_build_wt_kernel<<<(WT_ELEMS + 255) / 256, 256, 0, stream>>>(basis1, root1, wt1);
    rgcn_build_wt_kernel<<<(WT_ELEMS + 255) / 256, 256, 0, stream>>>(basis2, root2, wt2);

    // layer 1
    hipMemsetAsync(agg, 0, AGG_BYTES, stream);
    rgcn_scatter_kernel<<<N_EDGES, 256, 0, stream>>>(entity_emb, edge_index, edge_type,
                                                     cnt, comp1, agg);
    rgcn_gemm_kernel<<<GEMM_BLOCKS, 256, 0, stream>>>(agg, entity_emb, wt1, bias1, x1);

    // layer 2
    hipMemsetAsync(agg, 0, AGG_BYTES, stream);
    rgcn_scatter_kernel<<<N_EDGES, 256, 0, stream>>>(x1, edge_index, edge_type,
                                                     cnt, comp2, agg);
    rgcn_gemm_kernel<<<GEMM_BLOCKS, 256, 0, stream>>>(agg, x1, wt2, bias2, x2);

    // DistMult scoring
    rgcn_score_kernel<<<(N_TRIPLES * 32) / 256, 256, 0, stream>>>(x2, rel_emb, heads,
                                                                  rels, tails, out);
}
